// GLNN_76364518523597
// MI455X (gfx1250) — compile-verified, matrix operand order checked by reading, not run on hardware
//
#include <hip/hip_runtime.h>

typedef __attribute__((ext_vector_type(16))) _Float16 v16h;
typedef __attribute__((ext_vector_type(8)))  float    v8f;

#define EPS_BN 1e-5f
#define INV_MUL 0.17677669529663689f   /* 1/sqrt(32) */
#define INV_S3  0.57735026918962576f   /* 1/sqrt(3)  */

// K index for half-slot pair i (j=0/1) of a 16x32 f16 A fragment, lane group g.
__device__ __forceinline__ int kbase(int i, int g) {
  return ((i < 4) ? (2 * i) : (16 + 2 * (i - 4))) + g * 8;
}

// B fragment: columns [col0, col0+16) of a 32x32 row-major weight, scaled.
__device__ __forceinline__ v16h load_B_frag(const float* __restrict__ W, int col0, float scale) {
  const int lane = threadIdx.x & 31;
  const int g = lane >> 4, n = lane & 15;
  v16h b;
#pragma unroll
  for (int i = 0; i < 8; ++i) {
    const int k = kbase(i, g);
    b[2 * i]     = (_Float16)(W[(k)     * 32 + col0 + n] * scale);
    b[2 * i + 1] = (_Float16)(W[(k + 1) * 32 + col0 + n] * scale);
  }
  return b;
}

// Per-lane BN constants for this lane's 16 K slots.
// stats[0:32]=sum t0, [32:64]=sum t0^2, [64:96]=sum over nodes+comps of t1^2.
__device__ __forceinline__ void load_norm(const float* __restrict__ stats, float invN,
                                          float* __restrict__ mu, float* __restrict__ istd,
                                          float* __restrict__ inr) {
  const int lane = threadIdx.x & 31;
  const int g = lane >> 4;
#pragma unroll
  for (int i = 0; i < 8; ++i) {
    const int k = kbase(i, g);
#pragma unroll
    for (int j = 0; j < 2; ++j) {
      const int kk = k + j;
      const float m = stats[kk] * invN;
      mu[2 * i + j]   = m;
      istd[2 * i + j] = rsqrtf(stats[32 + kk] * invN - m * m + EPS_BN);
      inr[2 * i + j]  = rsqrtf(stats[64 + kk] * invN * (1.0f / 3.0f) + EPS_BN);
    }
  }
}

// Normalized 0e A fragment from one contiguous 32-float row.
__device__ __forceinline__ v16h load_A_row_norm(const float* __restrict__ row,
                                                const float* __restrict__ mu,
                                                const float* __restrict__ istd) {
  const int lane = threadIdx.x & 31;
  const int g = lane >> 4;
  v16h a;
#pragma unroll
  for (int i = 0; i < 8; ++i) {
    const int k = kbase(i, g);
    const float2 v = *(const float2*)(row + k);
    a[2 * i]     = (_Float16)((v.x - mu[2 * i])     * istd[2 * i]);
    a[2 * i + 1] = (_Float16)((v.y - mu[2 * i + 1]) * istd[2 * i + 1]);
  }
  return a;
}

// Normalized 1o A fragments (all 3 components) from one 96-float row ([32][3] interleave).
// Channel pair (k,k+1) spans 6 contiguous floats -> 3x float2 (8B aligned, k even).
__device__ __forceinline__ void load_A_row_vec3_norm(const float* __restrict__ row96,
                                                     const float* __restrict__ inr,
                                                     v16h& a0, v16h& a1, v16h& a2) {
  const int lane = threadIdx.x & 31;
  const int g = lane >> 4;
#pragma unroll
  for (int i = 0; i < 8; ++i) {
    const int k = kbase(i, g);
    const float* p = row96 + k * 3;
    const float2 q0 = *(const float2*)(p);
    const float2 q1 = *(const float2*)(p + 2);
    const float2 q2 = *(const float2*)(p + 4);
    const float s0 = inr[2 * i], s1 = inr[2 * i + 1];
    a0[2 * i]     = (_Float16)(q0.x * s0);
    a1[2 * i]     = (_Float16)(q0.y * s0);
    a2[2 * i]     = (_Float16)(q1.x * s0);
    a0[2 * i + 1] = (_Float16)(q1.y * s1);
    a1[2 * i + 1] = (_Float16)(q2.x * s1);
    a2[2 * i + 1] = (_Float16)(q2.y * s1);
  }
}

__device__ __forceinline__ v8f wmma_f16(v16h a, v16h b, v8f c) {
  return __builtin_amdgcn_wmma_f32_16x16x32_f16(false, a, false, b, (short)0, c, false, false);
}

// -------------------- utility kernels --------------------

__global__ void k_zero(float* __restrict__ p, int n) {
  int i = blockIdx.x * blockDim.x + threadIdx.x;
  if (i < n) p[i] = 0.0f;
}

// Identity-normalization stats for layer 0 (mu=0, istd=1, inr=1) + zero pool.
__global__ void k_init(float* __restrict__ stats, float* __restrict__ pool, int n) {
  const int i = threadIdx.x;
  if (i < 32)       stats[i] = 0.0f;
  else if (i < 64)  stats[i] = (float)n * (1.0f - EPS_BN);
  else if (i < 96)  stats[i] = 3.0f * (float)n * (1.0f - EPS_BN);
  else              pool[i - 96] = 0.0f;
}

// t0 = (x @ W_embed)/sqrt(5); t1 = 0   (layer-0 input state)
__global__ void k_embed(const float* __restrict__ x, const float* __restrict__ We,
                        float* __restrict__ t0, float* __restrict__ t1, int n) {
  const float s = 0.44721359549995793f; // 1/sqrt(5)
  const long long tot = (long long)n * 32;
  for (long long idx = (long long)blockIdx.x * blockDim.x + threadIdx.x; idx < tot;
       idx += (long long)gridDim.x * blockDim.x) {
    const int node = (int)(idx >> 5), c = (int)(idx & 31);
    float acc = 0.0f;
#pragma unroll
    for (int k = 0; k < 5; ++k) acc += x[node * 5 + k] * We[k * 32 + c];
    t0[idx] = acc * s;
    t1[idx * 3 + 0] = 0.0f;
    t1[idx * 3 + 1] = 0.0f;
    t1[idx * 3 + 2] = 0.0f;
  }
}

// -------------------- self-linear (WMMA, fused BN on gather) --------------------
// out0 = (bn(in0) @ Ws0)/sqrt(32) ; out1[:,:,c] = (bn(in1)[:,:,c] @ Ws1)/sqrt(32)
__global__ void k_self_linear(const float* __restrict__ in0, const float* __restrict__ in1,
                              const float* __restrict__ stats,
                              const float* __restrict__ Ws0, const float* __restrict__ Ws1,
                              float* __restrict__ out0, float* __restrict__ out1, int n) {
  const int lane = threadIdx.x & 31;
  const int g = lane >> 4, col = lane & 15, m = lane & 15;
  const v16h b0lo = load_B_frag(Ws0, 0, INV_MUL), b0hi = load_B_frag(Ws0, 16, INV_MUL);
  const v16h b1lo = load_B_frag(Ws1, 0, INV_MUL), b1hi = load_B_frag(Ws1, 16, INV_MUL);
  float mu[16], istd[16], inr[16];
  load_norm(stats, 1.0f / (float)n, mu, istd, inr);

  const int wave = threadIdx.x >> 5, wpb = blockDim.x >> 5;
  const int ntiles = n >> 4;
  for (int tile = blockIdx.x * wpb + wave; tile < ntiles; tile += gridDim.x * wpb) {
    const int base = tile << 4;
    // scalar (0e) block
    {
      const v16h a = load_A_row_norm(in0 + (size_t)(base + m) * 32, mu, istd);
      v8f clo = {}, chi = {};
      clo = wmma_f16(a, b0lo, clo);
      chi = wmma_f16(a, b0hi, chi);
#pragma unroll
      for (int v = 0; v < 8; ++v) {
        const size_t node = (size_t)(base + v + 8 * g);
        out0[node * 32 + col]      = clo[v];
        out0[node * 32 + col + 16] = chi[v];
      }
    }
    // vector (1o) block: 3 component GEMMs with shared B
    {
      v16h a0, a1, a2;
      load_A_row_vec3_norm(in1 + (size_t)(base + m) * 96, inr, a0, a1, a2);
      v8f c0lo = {}, c0hi = {}, c1lo = {}, c1hi = {}, c2lo = {}, c2hi = {};
      c0lo = wmma_f16(a0, b1lo, c0lo);
      c0hi = wmma_f16(a0, b1hi, c0hi);
      c1lo = wmma_f16(a1, b1lo, c1lo);
      c1hi = wmma_f16(a1, b1hi, c1hi);
      c2lo = wmma_f16(a2, b1lo, c2lo);
      c2hi = wmma_f16(a2, b1hi, c2hi);
#pragma unroll
      for (int v = 0; v < 8; ++v) {
        const size_t node = (size_t)(base + v + 8 * g);
        float* plo = out1 + (node * 32 + col) * 3;
        plo[0] = c0lo[v]; plo[1] = c1lo[v]; plo[2] = c2lo[v];
        float* phi = out1 + (node * 32 + col + 16) * 3;
        phi[0] = c0hi[v]; phi[1] = c1hi[v]; phi[2] = c2hi[v];
      }
    }
  }
}

// -------------------- edge messages (WMMA + scatter atomics, fused BN) --------------------
__global__ void k_edge(const float* __restrict__ in0, const float* __restrict__ in1,
                       const float* __restrict__ stats,
                       const int* __restrict__ src, const int* __restrict__ dst,
                       const float* __restrict__ ea,
                       const float* __restrict__ Wa, const float* __restrict__ Wb,
                       float* __restrict__ out0, float* __restrict__ out1, int e, int n) {
  const int lane = threadIdx.x & 31;
  const int g = lane >> 4, col = lane & 15, m = lane & 15;
  const v16h balo = load_B_frag(Wa, 0, INV_MUL), bahi = load_B_frag(Wa, 16, INV_MUL);
  const v16h bblo = load_B_frag(Wb, 0, INV_MUL), bbhi = load_B_frag(Wb, 16, INV_MUL);
  float mu[16], istd[16], inr[16];
  load_norm(stats, 1.0f / (float)n, mu, istd, inr);

  const int wave = threadIdx.x >> 5, wpb = blockDim.x >> 5;
  const int ntiles = e >> 4;
  for (int tile = blockIdx.x * wpb + wave; tile < ntiles; tile += gridDim.x * wpb) {
    const int e0 = tile << 4;
    const int s = src[e0 + m];

    // ---- m1 GEMM: bn(x0s) @ Wa
    const v16h a0 = load_A_row_norm(in0 + (size_t)s * 32, mu, istd);
    v8f m1lo = {}, m1hi = {};
    m1lo = wmma_f16(a0, balo, m1lo);
    m1hi = wmma_f16(a0, bahi, m1hi);

    // ---- d[e,u] = <bn(x1s)[e,u,:], ea[e,:]>/sqrt(3): linear -> fold inr into dot
    const float eax = ea[(size_t)(e0 + m) * 3 + 0];
    const float eay = ea[(size_t)(e0 + m) * 3 + 1];
    const float eaz = ea[(size_t)(e0 + m) * 3 + 2];
    v16h dfrag;
#pragma unroll
    for (int i = 0; i < 8; ++i) {
      const int k = kbase(i, g);
      const float* p = in1 + ((size_t)s * 32 + k) * 3;
      const float2 q0 = *(const float2*)(p);
      const float2 q1 = *(const float2*)(p + 2);
      const float2 q2 = *(const float2*)(p + 4);
      dfrag[2 * i]     = (_Float16)((q0.x * eax + q0.y * eay + q1.x * eaz) *
                                    (inr[2 * i] * INV_S3));
      dfrag[2 * i + 1] = (_Float16)((q1.y * eax + q2.x * eay + q2.y * eaz) *
                                    (inr[2 * i + 1] * INV_S3));
    }
    v8f m0lo = {}, m0hi = {};
    m0lo = wmma_f16(dfrag, bblo, m0lo);
    m0hi = wmma_f16(dfrag, bbhi, m0hi);

    // ---- scatter-add to destination nodes
#pragma unroll
    for (int v = 0; v < 8; ++v) {
      const int ee = e0 + v + 8 * g;
      const int d = dst[ee];
      const float ex = ea[(size_t)ee * 3 + 0];
      const float ey = ea[(size_t)ee * 3 + 1];
      const float ez = ea[(size_t)ee * 3 + 2];
      atomicAdd(&out0[(size_t)d * 32 + col],      m0lo[v]);
      atomicAdd(&out0[(size_t)d * 32 + col + 16], m0hi[v]);
      float* tv0 = &out1[((size_t)d * 32 + col) * 3];
      atomicAdd(tv0 + 0, m1lo[v] * ex);
      atomicAdd(tv0 + 1, m1lo[v] * ey);
      atomicAdd(tv0 + 2, m1lo[v] * ez);
      float* tv1 = &out1[((size_t)d * 32 + col + 16) * 3];
      atomicAdd(tv1 + 0, m1hi[v] * ex);
      atomicAdd(tv1 + 1, m1hi[v] * ey);
      atomicAdd(tv1 + 2, m1hi[v] * ez);
    }
  }
}

// -------------------- batch-norm statistics --------------------
__global__ void k_stats(const float* __restrict__ t0, const float* __restrict__ t1,
                        float* __restrict__ stats, int n) {
  float s0 = 0.0f, sq0 = 0.0f, sq1 = 0.0f;
  const long long tot = (long long)n * 32;
  const long long stride = (long long)gridDim.x * blockDim.x; // multiple of 32
  for (long long idx = (long long)blockIdx.x * blockDim.x + threadIdx.x; idx < tot; idx += stride) {
    const float v = t0[idx];
    s0 += v;
    sq0 += v * v;
    const float* p = t1 + idx * 3;
    const float2 q0 = *(const float2*)(p);
    const float z = p[2];
    sq1 += q0.x * q0.x + q0.y * q0.y + z * z;
  }
  __shared__ float sh[256];
  sh[threadIdx.x] = s0;
  __syncthreads();
  if (threadIdx.x < 32) {
    float a = 0.0f;
    for (int i = threadIdx.x; i < 256; i += 32) a += sh[i];
    atomicAdd(&stats[threadIdx.x], a);
  }
  __syncthreads();
  sh[threadIdx.x] = sq0;
  __syncthreads();
  if (threadIdx.x < 32) {
    float a = 0.0f;
    for (int i = threadIdx.x; i < 256; i += 32) a += sh[i];
    atomicAdd(&stats[32 + threadIdx.x], a);
  }
  __syncthreads();
  sh[threadIdx.x] = sq1;
  __syncthreads();
  if (threadIdx.x < 32) {
    float a = 0.0f;
    for (int i = threadIdx.x; i < 256; i += 32) a += sh[i];
    atomicAdd(&stats[64 + threadIdx.x], a);
  }
}

// -------------------- readout (BN applied on the fly) --------------------
__global__ void k_pool(const float* __restrict__ t0, const float* __restrict__ stats,
                       float* __restrict__ pool, int n) {
  const int c = threadIdx.x & 31;
  const float invN = 1.0f / (float)n;
  const float mu = stats[c] * invN;
  const float istd = rsqrtf(stats[32 + c] * invN - mu * mu + EPS_BN);
  float s = 0.0f;
  const long long tot = (long long)n * 32;
  const long long stride = (long long)gridDim.x * blockDim.x;
  for (long long idx = (long long)blockIdx.x * blockDim.x + threadIdx.x; idx < tot; idx += stride)
    s += (t0[idx] - mu) * istd;
  __shared__ float sh[256];
  sh[threadIdx.x] = s;
  __syncthreads();
  if (threadIdx.x < 32) {
    float a = 0.0f;
    for (int i = threadIdx.x; i < 256; i += 32) a += sh[i];
    atomicAdd(&pool[threadIdx.x], a);
  }
}

__global__ void k_finish(const float* __restrict__ pool, const float* __restrict__ Wr,
                         float* __restrict__ out, int n) {
  if (threadIdx.x == 0) {
    float s = 0.0f;
    for (int c = 0; c < 32; ++c) s += (pool[c] / (float)n) * Wr[c];
    out[0] = s * INV_MUL;
  }
}

// -------------------- driver --------------------
extern "C" void kernel_launch(void* const* d_in, const int* in_sizes, int n_in,
                              void* d_out, int out_size, void* d_ws, size_t ws_size,
                              hipStream_t stream) {
  (void)n_in; (void)out_size; (void)ws_size;
  const float* x   = (const float*)d_in[0];
  const int*   ei  = (const int*)d_in[1];
  const float* ea  = (const float*)d_in[2];
  const float* We  = (const float*)d_in[3];
  const float* Ws0 = (const float*)d_in[4];
  const float* Ws1 = (const float*)d_in[5];
  const float* Wa  = (const float*)d_in[6];
  const float* Wb  = (const float*)d_in[7];
  const float* Wr  = (const float*)d_in[8];

  const int n = in_sizes[0] / 5;
  const int e = in_sizes[1] / 2;
  const int* src = ei;
  const int* dst = ei + e;

  float* ws = (float*)d_ws;
  float* t0buf[2];
  float* t1buf[2];
  t0buf[0] = ws;                          // n*32
  t1buf[0] = t0buf[0] + (size_t)n * 32;   // n*96
  t0buf[1] = t1buf[0] + (size_t)n * 96;   // n*32
  t1buf[1] = t0buf[1] + (size_t)n * 32;   // n*96
  float* statsbuf[2];
  statsbuf[0] = t1buf[1] + (size_t)n * 96; // 128
  statsbuf[1] = statsbuf[0] + 128;         // 128
  float* pool = statsbuf[1] + 128;         // 32

  // layer-0 identity stats + zero readout accumulator
  k_init<<<1, 128, 0, stream>>>(statsbuf[0], pool, n);

  k_embed<<<4096, 256, 0, stream>>>(x, We, t0buf[0], t1buf[0], n);

  const int wpb = 8; // waves per 256-thread block
  const int grid_sl = ((n >> 4) + wpb - 1) / wpb;
  const int grid_eg = ((e >> 4) + wpb - 1) / wpb;

  int cur = 0;
  for (int l = 0; l < 3; ++l) {
    const int nxt = cur ^ 1;
    k_zero<<<1, 128, 0, stream>>>(statsbuf[nxt], 96);
    k_self_linear<<<grid_sl, 256, 0, stream>>>(t0buf[cur], t1buf[cur], statsbuf[cur],
                                               Ws0 + l * 1024, Ws1 + l * 1024,
                                               t0buf[nxt], t1buf[nxt], n);
    k_edge<<<grid_eg, 256, 0, stream>>>(t0buf[cur], t1buf[cur], statsbuf[cur],
                                        src, dst, ea, Wa + l * 1024, Wb + l * 1024,
                                        t0buf[nxt], t1buf[nxt], e, n);
    k_stats<<<1024, 256, 0, stream>>>(t0buf[nxt], t1buf[nxt], statsbuf[nxt], n);
    cur = nxt;
  }

  k_pool<<<1024, 256, 0, stream>>>(t0buf[cur], statsbuf[cur], pool, n);
  k_finish<<<1, 32, 0, stream>>>(pool, Wr, (float*)d_out, n);
}
